// Multilevel_Fusion_Transformer_63866163692031
// MI455X (gfx1250) — matrix operand order repaired and checked by reading, not compiled, not run on hardware
//
#include <hip/hip_runtime.h>
#include <hip/hip_bf16.h>
#include <stddef.h>
#include <stdint.h>

// ---------------- CDNA5 / gfx1250 types ----------------
typedef _Float16 v16h __attribute__((ext_vector_type(16)));
typedef _Float16 v8h  __attribute__((ext_vector_type(8)));
typedef float    v8f  __attribute__((ext_vector_type(8)));
typedef unsigned int u32x4 __attribute__((ext_vector_type(4)));
typedef int          i32x4 __attribute__((ext_vector_type(4)));
typedef int          i32x8 __attribute__((ext_vector_type(8)));

#define BATCH   32
#define CDIM    256
#define HW      4096
#define ROWS    (BATCH * CDIM)   // 8192
#define NH      5
#define LD      10
#define PROJ    50               // NH*LD
#define PP      64               // padded proj (N for proj GEMM, K for fc GEMM)
#define FEAT    (CDIM * LD)      // 2560
#define BN_N    (BATCH * HW)     // 131072 elems per channel

// LDS tile for gemm_proj: 128 rows x 32 f32, padded 16B per 128B row
// -> row stride 36 DWORDs (144B, 16B aligned, conflict-free b128 reads).
#define XROW_DW   36
#define XBUF_DW   (128 * XROW_DW)       // 4608 f32 per buffer
#define XBUF_B    (XBUF_DW * 4)         // 18432 bytes

// ============================================================
// Tensor Data Mover: DMA a 128x32 f32 tile (row stride 4096 elems)
// from global into LDS with 16B padding every 128B row.
// D# layout per CDNA5 ISA.
// ============================================================
__device__ __forceinline__ void tdm_load_tile(const float* gaddr, unsigned lds_off) {
  const unsigned long long ga = (unsigned long long)(uintptr_t)gaddr;
  u32x4 g0;
  g0[0] = 1u;                                              // count=1 (user mode)
  g0[1] = lds_off;                                         // lds_addr (bytes)
  g0[2] = (unsigned)(ga & 0xFFFFFFFFull);                  // global_addr lo
  g0[3] = (unsigned)((ga >> 32) & 0x1FFFFFFull) | (2u << 30); // addr hi | type=2

  i32x8 g1;
  g1[0] = (2 << 16)            // data_size = 4 bytes
        | (1 << 20)            // pad_enable
        | (4 << 22)            // pad_interval: 1<<4 x 8B = 128B
        | (3 << 25);           // pad_amount: 4 DWORDs = 16B
  g1[1] = (int)((HW & 0xFFFF) << 16);            // tensor_dim0 lo16 (4096)
  g1[2] = (int)((ROWS & 0xFFFF) << 16);          // dim0 hi16=0 | tensor_dim1 lo16 (8192)
  g1[3] = (int)(32u << 16);                      // dim1 hi16=0 | tile_dim0 = 32
  g1[4] = 128;                                   // tile_dim1 = 128 | tile_dim2 = 0
  g1[5] = HW;                                    // tensor_dim0_stride lo32 = 4096
  g1[6] = 0;                                     // stride hi16 | tensor_dim1_stride lo16
  g1[7] = 0;                                     // tensor_dim1_stride hi32
  const i32x4 z4 = {0, 0, 0, 0};
#if defined(__clang_major__) && __clang_major__ >= 23
  const i32x8 z8 = {0, 0, 0, 0, 0, 0, 0, 0};
  __builtin_amdgcn_tensor_load_to_lds(g0, g1, z4, z4, z8, 0);
#else
  __builtin_amdgcn_tensor_load_to_lds(g0, g1, z4, z4, 0);
#endif
}

// ============================================================
// Weight packing into WMMA B-fragment order (f32 -> f16).
// B 32x16 f16 layout: lane n (0..15) col = n, elem i -> K = k0+(i<8 ? i : 8+i);
//                     lane n+16      col = n, elem i -> K = k0+8+(i<8 ? i : 8+i).
// ============================================================
__global__ __launch_bounds__(256) void prep_wproj(const float* __restrict__ w,
                                                  _Float16* __restrict__ wf) {
  // wf[((s*4+t)*32+lane)*16+i], s:128 kstep, t:4 ntile  -> 262144 halves
  int idx  = blockIdx.x * 256 + threadIdx.x;
  int i    = idx & 15;
  int lane = (idx >> 4) & 31;
  int t    = (idx >> 9) & 3;
  int s    = idx >> 11;
  int k    = s * 32 + (i & 7) + ((i & 8) << 1) + ((lane & 16) >> 1);
  int col  = t * 16 + (lane & 15);
  float v  = (col < PROJ) ? w[k * PROJ + col] : 0.0f;
  wf[idx]  = (_Float16)v;
}

__global__ __launch_bounds__(256) void prep_wfc(const float* __restrict__ w,
                                                _Float16* __restrict__ wf) {
  // wf[((s*256+t)*32+lane)*16+i], s:2 kstep, t:256 ntile -> 262144 halves
  int idx  = blockIdx.x * 256 + threadIdx.x;
  int i    = idx & 15;
  int lane = (idx >> 4) & 31;
  int t    = (idx >> 9) & 255;
  int s    = idx >> 17;
  int k    = s * 32 + (i & 7) + ((i & 8) << 1) + ((lane & 16) >> 1);
  int col  = t * 16 + (lane & 15);
  float v  = (k < PROJ) ? w[k * HW + col] : 0.0f;   // zero-pad K rows 50..63
  wf[idx]  = (_Float16)v;
}

// Build an A-fragment (16-bit A 16x32 layout) from 32 consecutive f32 of a row;
// caller pre-applies the lane-half offset (0 or 8).
__device__ __forceinline__ v16h afrag_from_f32(const float* __restrict__ p) {
  v16h a;
#pragma unroll
  for (int i = 0; i < 8; ++i) a[i] = (_Float16)p[i];
#pragma unroll
  for (int i = 0; i < 8; ++i) a[8 + i] = (_Float16)p[16 + i];
  return a;
}

// ============================================================
// Projection GEMM: P[8192,64] = f16(X[8192,4096]) @ f16(W[4096,64])
// Block: 256 thr (8 waves). Block tile = 128 rows; wave tile = 16x64.
// X staged by the Tensor Data Mover (double-buffered LDS), f32 in LDS,
// converted to f16 while building A fragments.
// ============================================================
__global__ __launch_bounds__(256) void gemm_proj(const float* __restrict__ X,
                                                 const _Float16* __restrict__ Wf,
                                                 float* __restrict__ P) {
  __shared__ float sX[2 * XBUF_DW];
  const int wave = threadIdx.x >> 5;
  const int lane = threadIdx.x & 31;
  const int m0b  = blockIdx.x * 128;
  const int mw   = wave * 16;
  const int lrow = mw + (lane & 15);
  const int aoff = (lane & 16) ? 8 : 0;
  const unsigned lds_base = (unsigned)(uintptr_t)&sX[0];

  v8f c0 = {}, c1 = {}, c2 = {}, c3 = {};

  // prologue: TDM fetch of tile s=0 into buffer 0 (one op per block)
  if (wave == 0) {
    tdm_load_tile(X + (size_t)m0b * HW, lds_base);
  }

  for (int s = 0; s < HW / 32; ++s) {
    __syncthreads();   // everyone done reading the buffer the next TDM overwrites
    if (wave == 0) {
      if (s + 1 < HW / 32) {
        tdm_load_tile(X + (size_t)m0b * HW + (s + 1) * 32,
                      lds_base + (unsigned)(((s + 1) & 1) * XBUF_B));
        __builtin_amdgcn_s_wait_tensorcnt(1);  // tile s complete (in-order TDM)
      } else {
        __builtin_amdgcn_s_wait_tensorcnt(0);
      }
    }
    __syncthreads();   // publish tile s to all waves

    // A fragment: f32 from LDS, convert to f16 (cvt co-executes with WMMA)
    const float* ap = &sX[(s & 1) * XBUF_DW + lrow * XROW_DW + aoff];
    const v16h a = afrag_from_f32(ap);

    // B fragments: fully coalesced 32B/lane from packed weights (L2-hot)
    const _Float16* wb = Wf + ((size_t)(s * 4) * 32 + lane) * 16;
    v16h b0 = *(const v16h*)(wb + 0 * 512);
    v16h b1 = *(const v16h*)(wb + 1 * 512);
    v16h b2 = *(const v16h*)(wb + 2 * 512);
    v16h b3 = *(const v16h*)(wb + 3 * 512);

    c0 = __builtin_amdgcn_wmma_f32_16x16x32_f16(false, a, false, b0, (short)0, c0, false, false);
    c1 = __builtin_amdgcn_wmma_f32_16x16x32_f16(false, a, false, b1, (short)0, c1, false, false);
    c2 = __builtin_amdgcn_wmma_f32_16x16x32_f16(false, a, false, b2, (short)0, c2, false, false);
    c3 = __builtin_amdgcn_wmma_f32_16x16x32_f16(false, a, false, b3, (short)0, c3, false, false);
  }

  // C layout: VGPR r -> M = r (+8 for high lanes), N = lane%16 + 16*tile
  const int nlo = lane & 15;
  const int mhi = (lane >> 4) * 8;
#pragma unroll
  for (int r = 0; r < 8; ++r) {
    float* o = P + (size_t)(m0b + mw + mhi + r) * PP + nlo;
    o[0]  = c0[r];
    o[16] = c1[r];
    o[32] = c2[r];
    o[48] = c3[r];
  }
}

// ============================================================
// Attention over heads (per batch): 5x5 softmax, residual inside.
// Qf[b,i,c*10+l] = Pq[(b*256+c)*64 + i*10 + l]
// ============================================================
__global__ __launch_bounds__(256) void attn_kernel(const float* __restrict__ Pq,
                                                   const float* __restrict__ Pk,
                                                   const float* __restrict__ Pv,
                                                   float* __restrict__ Va) {
  const int b   = blockIdx.x;
  const int tid = threadIdx.x;
  __shared__ float sS[NH * NH];
  __shared__ float sA[NH * NH];
  if (tid < NH * NH) sS[tid] = 0.0f;
  __syncthreads();

  float acc[NH][NH];
#pragma unroll
  for (int i = 0; i < NH; ++i)
#pragma unroll
    for (int j = 0; j < NH; ++j) acc[i][j] = 0.0f;

  for (int cl = tid; cl < FEAT; cl += 256) {
    const int c = cl / LD, l = cl % LD;
    const size_t base = ((size_t)(b * CDIM + c)) * PP + l;
    float qv[NH], kv[NH];
#pragma unroll
    for (int i = 0; i < NH; ++i) { qv[i] = Pq[base + i * LD]; kv[i] = Pk[base + i * LD]; }
#pragma unroll
    for (int i = 0; i < NH; ++i)
#pragma unroll
      for (int j = 0; j < NH; ++j) acc[i][j] += qv[i] * kv[j];
  }
#pragma unroll
  for (int i = 0; i < NH; ++i)
#pragma unroll
    for (int j = 0; j < NH; ++j) atomicAdd(&sS[i * NH + j], acc[i][j]);
  __syncthreads();

  if (tid < NH) {
    const float inv_scale = 1.0f / 50.596442562694074f;  // 1/sqrt(2560)
    float row[NH], m = -1e30f;
#pragma unroll
    for (int j = 0; j < NH; ++j) { row[j] = sS[tid * NH + j] * inv_scale; m = fmaxf(m, row[j]); }
    float sum = 0.0f;
#pragma unroll
    for (int j = 0; j < NH; ++j) { row[j] = __expf(row[j] - m); sum += row[j]; }
    const float inv = 1.0f / sum;
#pragma unroll
    for (int j = 0; j < NH; ++j) sA[tid * NH + j] = row[j] * inv;
  }
  __syncthreads();

  for (int cl = tid; cl < FEAT; cl += 256) {
    const int c = cl / LD, l = cl % LD;
    const size_t base = ((size_t)(b * CDIM + c)) * PP + l;
    float vv[NH];
#pragma unroll
    for (int j = 0; j < NH; ++j) vv[j] = Pv[base + j * LD];
#pragma unroll
    for (int i = 0; i < NH; ++i) {
      float s = vv[i];  // residual inside attention
#pragma unroll
      for (int j = 0; j < NH; ++j) s += sA[i * NH + j] * vv[j];
      Va[base + i * LD] = s;
    }
  }
  // zero K-pad columns 50..63 so the padded fc GEMM reads exact zeros
  for (int p = tid; p < CDIM * (PP - PROJ); p += 256) {
    int c = p / (PP - PROJ), col = PROJ + p % (PP - PROJ);
    Va[((size_t)(b * CDIM + c)) * PP + col] = 0.0f;
  }
}

// ============================================================
// fc GEMM + residual + store + per-channel stats:
// Out[8192,4096] = Vin + f16(Va[8192,64]) @ f16(Wfc[64,4096])
// Wave tile 16 rows x 64 cols, swept over N; A frags loop-invariant.
// ============================================================
__global__ __launch_bounds__(256) void gemm_fc(const float* __restrict__ Va,
                                               const _Float16* __restrict__ Wf,
                                               const float* __restrict__ Vin,
                                               float* __restrict__ Out,
                                               float* __restrict__ csum,
                                               float* __restrict__ csumsq) {
  const int wave = threadIdx.x >> 5;
  const int lane = threadIdx.x & 31;
  const int m0   = (blockIdx.x * 8 + wave) * 16;
  const int aoff = (lane & 16) ? 8 : 0;
  const int row_a = m0 + (lane & 15);

  const float* ar = Va + (size_t)row_a * PP;
  const v16h a0 = afrag_from_f32(ar + aoff);        // K 0..31
  const v16h a1 = afrag_from_f32(ar + 32 + aoff);   // K 32..63

  const int nlo = lane & 15;
  const int mhi = (lane >> 4) * 8;
  float sumr[8], sumsqr[8];
#pragma unroll
  for (int r = 0; r < 8; ++r) { sumr[r] = 0.0f; sumsqr[r] = 0.0f; }

  for (int n0 = 0; n0 < HW; n0 += 64) {
    const int t0 = n0 >> 4;
    v8f c[4] = {};
#pragma unroll
    for (int t = 0; t < 4; ++t) {
      const _Float16* wb0 = Wf + (((size_t)(t0 + t)) * 32 + lane) * 16;
      const _Float16* wb1 = Wf + (((size_t)(256 + t0 + t)) * 32 + lane) * 16;
      v16h b0 = *(const v16h*)wb0;
      v16h b1 = *(const v16h*)wb1;
      c[t] = __builtin_amdgcn_wmma_f32_16x16x32_f16(false, a0, false, b0, (short)0, c[t], false, false);
      c[t] = __builtin_amdgcn_wmma_f32_16x16x32_f16(false, a1, false, b1, (short)0, c[t], false, false);
    }
#pragma unroll
    for (int r = 0; r < 8; ++r) {
      const size_t rb = (size_t)(m0 + mhi + r) * HW + n0 + nlo;
#pragma unroll
      for (int t = 0; t < 4; ++t) {
        float v = c[t][r] + Vin[rb + t * 16];
        Out[rb + t * 16] = v;
        sumr[r] += v;
        sumsqr[r] += v * v;
      }
    }
  }

  // reduce across the 16-lane half (lanes of one half share the row)
#pragma unroll
  for (int r = 0; r < 8; ++r) {
    float s = sumr[r], q = sumsqr[r];
#pragma unroll
    for (int m = 1; m <= 8; m <<= 1) {
      s += __shfl_xor(s, m, 32);
      q += __shfl_xor(q, m, 32);
    }
    if ((lane & 15) == 0) {
      const int ch = (m0 + mhi + r) & (CDIM - 1);   // row % 256 == channel
      atomicAdd(&csum[ch], s);
      atomicAdd(&csumsq[ch], q);
    }
  }
}

__global__ void zero_stats(float* csum, float* csumsq) {
  csum[threadIdx.x] = 0.0f;
  csumsq[threadIdx.x] = 0.0f;
}

__global__ void finalize_stats(const float* __restrict__ csum,
                               const float* __restrict__ csumsq,
                               const float* __restrict__ gamma,
                               const float* __restrict__ beta,
                               float* __restrict__ scl,
                               float* __restrict__ sht) {
  const int c = threadIdx.x;
  const float inv_n = 1.0f / (float)BN_N;
  const float mean = csum[c] * inv_n;
  const float var  = csumsq[c] * inv_n - mean * mean;
  const float inv  = rsqrtf(var + 1e-5f);
  const float g    = gamma[c] * inv;
  scl[c] = g;
  sht[c] = beta[c] - mean * g;
}

__global__ __launch_bounds__(256) void bn_apply(float* __restrict__ Out,
                                                const float* __restrict__ scl,
                                                const float* __restrict__ sht) {
  const size_t i4 = (size_t)blockIdx.x * 256 + threadIdx.x;  // float4 index
  const int ch = (int)((i4 >> 10) & (CDIM - 1));             // (i4*4/4096)%256
  const float a = scl[ch], b = sht[ch];
  float4* p = (float4*)Out + i4;
  float4 v = *p;
  v.x = v.x * a + b;
  v.y = v.y * a + b;
  v.z = v.z * a + b;
  v.w = v.w * a + b;
  *p = v;
}

// ============================================================
extern "C" void kernel_launch(void* const* d_in, const int* in_sizes, int n_in,
                              void* d_out, int out_size, void* d_ws, size_t ws_size,
                              hipStream_t stream) {
  const float* q     = (const float*)d_in[0];
  const float* k     = (const float*)d_in[1];
  const float* v     = (const float*)d_in[2];
  const float* w_q   = (const float*)d_in[3];
  const float* w_k   = (const float*)d_in[4];
  const float* w_v   = (const float*)d_in[5];
  const float* w_fc  = (const float*)d_in[6];
  const float* gamma = (const float*)d_in[7];
  const float* beta  = (const float*)d_in[8];
  float* out = (float*)d_out;

  // workspace layout (~10 MB)
  _Float16* wqf = (_Float16*)d_ws;            // 262144 halves each
  _Float16* wkf = wqf + 262144;
  _Float16* wvf = wkf + 262144;
  _Float16* wff = wvf + 262144;
  float* Pq   = (float*)(wff + 262144);       // 524288 f32 each ([8192,64])
  float* Pk   = Pq + ROWS * PP;
  float* Pv   = Pk + ROWS * PP;
  float* Vab  = Pv + ROWS * PP;
  float* csum   = Vab + ROWS * PP;            // 256
  float* csumsq = csum + CDIM;                // 256
  float* scl    = csumsq + CDIM;              // 256
  float* sht    = scl + CDIM;                 // 256

  zero_stats<<<1, 256, 0, stream>>>(csum, csumsq);
  prep_wproj<<<1024, 256, 0, stream>>>(w_q, wqf);
  prep_wproj<<<1024, 256, 0, stream>>>(w_k, wkf);
  prep_wproj<<<1024, 256, 0, stream>>>(w_v, wvf);
  prep_wfc<<<1024, 256, 0, stream>>>(w_fc, wff);

  gemm_proj<<<ROWS / 128, 256, 0, stream>>>(q, wqf, Pq);
  gemm_proj<<<ROWS / 128, 256, 0, stream>>>(k, wkf, Pk);
  gemm_proj<<<ROWS / 128, 256, 0, stream>>>(v, wvf, Pv);

  attn_kernel<<<BATCH, 256, 0, stream>>>(Pq, Pk, Pv, Vab);

  gemm_fc<<<ROWS / 128, 256, 0, stream>>>(Vab, wff, v, out, csum, csumsq);
  finalize_stats<<<1, 256, 0, stream>>>(csum, csumsq, gamma, beta, scl, sht);
  bn_apply<<<(ROWS * HW / 4) / 256, 256, 0, stream>>>(out, scl, sht);
}
